// MOELayer_24653112279122
// MI455X (gfx1250) — compile-verified, matrix operand order checked by reading, not run on hardware
//
#include <hip/hip_runtime.h>
#include <hip/hip_bf16.h>
#include <float.h>
#include <stdint.h>

// Problem constants (from reference)
#define S_TOK 8192
#define MDIM  1024
#define NEXP  8
#define HDIM  4096
#define CAPN  2048   // 2 * ceil(S/E)

typedef __attribute__((ext_vector_type(16))) __bf16 v16bf;
typedef __attribute__((ext_vector_type(8)))  float  v8f;

union Frag16 { v16bf v; unsigned int u[8]; };

// CDNA5 async global->LDS direct copy (ASYNCcnt-tracked). Probe-verified on
// this toolchain: params are pointers to 4xi32 vectors (AS1 global/AS3 lds).
#if defined(__gfx1250__) && __has_builtin(__builtin_amdgcn_global_load_async_to_lds_b128)
#define USE_ASYNC_LDS 1
typedef __attribute__((ext_vector_type(4))) int v4i;
typedef __attribute__((address_space(1))) v4i as1_v4i;
typedef __attribute__((address_space(3))) v4i as3_v4i;
#else
#define USE_ASYNC_LDS 0
#endif

__device__ __forceinline__ void wait_async0() {
#if USE_ASYNC_LDS
#if __has_builtin(__builtin_amdgcn_s_wait_asynccnt)
    __builtin_amdgcn_s_wait_asynccnt(0);
#else
    asm volatile("s_wait_asynccnt 0" ::: "memory");
#endif
#endif
}

// Hardware bf16 conversion (RNE) — lets the compiler use native cvt ops
// instead of integer bit-twiddling.
__device__ __forceinline__ unsigned short f2bf(float f) {
    __bf16 h = (__bf16)f;
    return __builtin_bit_cast(unsigned short, h);
}
__device__ __forceinline__ unsigned int pack_bf16(float lo, float hi) {
    return (unsigned int)f2bf(lo) | ((unsigned int)f2bf(hi) << 16);
}

// ---------------------------------------------------------------------------
// Zero-init
// ---------------------------------------------------------------------------
__global__ void k_init(unsigned int* __restrict__ p, long n) {
    long i = (long)blockIdx.x * blockDim.x + threadIdx.x;
    long stride = (long)gridDim.x * blockDim.x;
    for (; i < n; i += stride) p[i] = 0u;
}

// ---------------------------------------------------------------------------
// Gating: one wave per token; gate weights staged in LDS; deterministic
// per-block partial me sums (no float atomics).
// ---------------------------------------------------------------------------
__global__ __launch_bounds__(256) void k_gate(
    const float* __restrict__ x, const float* __restrict__ wg,
    int* __restrict__ idx1, int* __restrict__ idx2,
    float* __restrict__ g1o, float* __restrict__ g2o,
    float* __restrict__ partial_me)
{
    __shared__ float lw[NEXP * MDIM];
    __shared__ float lg[8][NEXP];
    const int tid = threadIdx.x;

    {
        const float4* s = (const float4*)wg;
        float4* d = (float4*)lw;
        #pragma unroll
        for (int i = 0; i < (NEXP * MDIM) / 4 / 256; ++i)
            d[tid + i * 256] = s[tid + i * 256];
    }
    __syncthreads();

    const int wid  = tid >> 5;
    const int lane = tid & 31;
    const int t = blockIdx.x * 8 + wid;

    float acc[NEXP] = {};
    const float4* xr  = (const float4*)(x + (size_t)t * MDIM);
    const float4* lw4 = (const float4*)lw;
    #pragma unroll
    for (int i = 0; i < 8; ++i) {
        float4 xv = xr[lane + i * 32];
        #pragma unroll
        for (int e = 0; e < NEXP; ++e) {
            float4 wv = lw4[e * 256 + lane + i * 32];
            acc[e] = fmaf(xv.x, wv.x, fmaf(xv.y, wv.y,
                     fmaf(xv.z, wv.z, fmaf(xv.w, wv.w, acc[e]))));
        }
    }
    #pragma unroll
    for (int e = 0; e < NEXP; ++e)
        #pragma unroll
        for (int m = 16; m > 0; m >>= 1)
            acc[e] += __shfl_xor(acc[e], m, 32);

    if (lane == 0) {
        float mx = acc[0];
        #pragma unroll
        for (int e = 1; e < NEXP; ++e) mx = fmaxf(mx, acc[e]);
        float g[NEXP]; float den = 0.f;
        #pragma unroll
        for (int e = 0; e < NEXP; ++e) { g[e] = __expf(acc[e] - mx); den += g[e]; }
        float inv = 1.f / den;
        #pragma unroll
        for (int e = 0; e < NEXP; ++e) g[e] *= inv;

        int i1 = 0;
        #pragma unroll
        for (int e = 1; e < NEXP; ++e) if (acc[e] > acc[i1]) i1 = e;
        int i2 = (i1 == 0) ? 1 : 0;
        #pragma unroll
        for (int e = 0; e < NEXP; ++e) if (e != i1 && acc[e] > acc[i2]) i2 = e;

        float a = g[i1], b = g[i2];
        float den2 = fmaxf(a + b, 1.1920929e-7f);
        idx1[t] = i1; idx2[t] = i2;
        g1o[t] = a / den2; g2o[t] = b / den2;
        #pragma unroll
        for (int e = 0; e < NEXP; ++e) lg[wid][e] = g[e];
    }
    __syncthreads();
    if (tid < NEXP) {
        float s = 0.f;
        #pragma unroll
        for (int w = 0; w < 8; ++w) s += lg[w][tid];
        partial_me[blockIdx.x * NEXP + tid] = s;
    }
}

// ---------------------------------------------------------------------------
// Ordered capacity scan (reference cumsum semantics); trivial serial work.
// ---------------------------------------------------------------------------
__global__ void k_loc(const int* __restrict__ idx1, const int* __restrict__ idx2,
                      int* __restrict__ loc1, int* __restrict__ loc2,
                      int* __restrict__ ce)
{
    int c1[NEXP] = {};
    for (int t = 0; t < S_TOK; ++t) loc1[t] = c1[idx1[t]]++;
    for (int e = 0; e < NEXP; ++e) ce[e] = c1[e];
    int c2[NEXP] = {};
    for (int t = 0; t < S_TOK; ++t) { int e = idx2[t]; loc2[t] = c1[e] + c2[e]++; }
}

// ---------------------------------------------------------------------------
// Dispatch: scatter token rows into disp[E,CAP,M] as bf16 (unique slots).
// ---------------------------------------------------------------------------
__global__ __launch_bounds__(256) void k_dispatch(
    const float* __restrict__ x,
    const int* __restrict__ idx1, const int* __restrict__ idx2,
    const int* __restrict__ loc1, const int* __restrict__ loc2,
    unsigned short* __restrict__ disp)
{
    const int t = blockIdx.x;
    const int col = threadIdx.x * 4;
    float4 xv = *(const float4*)(x + (size_t)t * MDIM + col);
    unsigned int p0 = pack_bf16(xv.x, xv.y);
    unsigned int p1 = pack_bf16(xv.z, xv.w);

    int e1 = idx1[t], l1 = loc1[t];
    if (l1 < CAPN) {
        unsigned int* d = (unsigned int*)(disp + ((size_t)e1 * CAPN + l1) * MDIM + col);
        d[0] = p0; d[1] = p1;
    }
    int e2 = idx2[t], l2 = loc2[t];
    if (l2 < CAPN) {
        unsigned int* d = (unsigned int*)(disp + ((size_t)e2 * CAPN + l2) * MDIM + col);
        d[0] = p0; d[1] = p1;
    }
}

// ---------------------------------------------------------------------------
// WMMA fragment load from LDS. 16-bit matrix 16x32 layout (ISA 7.12.2):
// lane&15 = row; lanes<16 hold K {0..7,16..23}; lanes>=16 hold K {8..15,24..31};
// each dword holds the (2k,2k+1) bf16 pair.
// ---------------------------------------------------------------------------
__device__ __forceinline__ void load_frag(const unsigned int* __restrict__ lds,
                                          int row, int strideDw, int lane,
                                          unsigned int out[8])
{
    const int half = (lane >> 4) & 1;
    const unsigned int* base = lds + (row + (lane & 15)) * strideDw;
    #pragma unroll
    for (int j = 0; j < 4; ++j) {
        out[j]     = base[j + half * 4];       // K = 0..15 region
        out[j + 4] = base[j + half * 4 + 8];   // K = 16..31 region
    }
}

// ---------------------------------------------------------------------------
// Expert GEMM: C = A(bf16) x B(f32->bf16) + bias. Block tile 128x128,
// 8 waves, wave tile 32x64 = 2x4 WMMA 16x16 tiles, K-tile 32.
// Double-buffered LDS; A staged with async global->LDS DMA (ASYNCcnt) and
// B(kb+1) register-staged, both overlapping the WMMA block of kb.
// ---------------------------------------------------------------------------
#define LDA_DW 28   // 112B row stride: 16B-aligned for async b128, 16 banks
#define LDB_DW 25   // 100B row stride: gcd(25,64)=1 -> conflict-free frag reads

template <bool RELU_BF16>
__global__ __launch_bounds__(256) void k_gemm(
    const unsigned short* __restrict__ Aall, const float* __restrict__ Ball,
    const float* __restrict__ biasall, void* __restrict__ Outall,
    int K, int N, int tilesN)
{
    __shared__ unsigned int lA[2][128 * LDA_DW];
    __shared__ unsigned int lB[2][128 * LDB_DW];

    const int e = blockIdx.y;
    const unsigned short* A = Aall + (size_t)e * CAPN * K;
    const float* B    = Ball    + (size_t)e * K * N;
    const float* bias = biasall + (size_t)e * N;

    const int tm = blockIdx.x / tilesN;
    const int tn = blockIdx.x % tilesN;
    const int tid  = threadIdx.x;
    const int lane = tid & 31;
    const int wid  = tid >> 5;
    const int wm = (wid & 3) * 32;
    const int wn = (wid >> 2) * 64;

    v8f c[2][4] = {};

    // A-copy mapping: thread -> (row, 16-element half)
    const int arow  = tid >> 1;
    const int ahalf = tid & 1;
    // B-copy mapping: thread -> cols n0..n0+7 of k-rows (k0, k0+1)
    const int bn0 = (tid & 15) * 8;
    const int bk0 = (tid >> 4) * 2;
    const int kd  = bk0 >> 1;

    // Register staging for the next B tile.
    float4 r0a, r0b, r1a, r1b;

#define ISSUE_A(kb_, buf_)                                                     \
    do {                                                                       \
        const unsigned short* gsrc_ = A + (size_t)(tm * 128 + arow) * K        \
                                      + (kb_) * 32 + ahalf * 16;               \
        unsigned int* ldst_ = &lA[buf_][arow * LDA_DW + ahalf * 8];            \
        ASYNC_COPY(gsrc_, ldst_);                                              \
    } while (0)

#if USE_ASYNC_LDS
#define ASYNC_COPY(gsrc_, ldst_)                                               \
    do {                                                                       \
        __builtin_amdgcn_global_load_async_to_lds_b128(                        \
            (as1_v4i*)(uintptr_t)(gsrc_), (as3_v4i*)(uintptr_t)(ldst_), 0, 0); \
        __builtin_amdgcn_global_load_async_to_lds_b128(                        \
            (as1_v4i*)(uintptr_t)(gsrc_), (as3_v4i*)(uintptr_t)(ldst_), 16, 0);\
    } while (0)
#else
#define ASYNC_COPY(gsrc_, ldst_)                                               \
    do {                                                                       \
        const uint4* s4_ = (const uint4*)(gsrc_);                              \
        uint4 p0_ = s4_[0], p1_ = s4_[1];                                      \
        (ldst_)[0] = p0_.x; (ldst_)[1] = p0_.y;                                \
        (ldst_)[2] = p0_.z; (ldst_)[3] = p0_.w;                                \
        (ldst_)[4] = p1_.x; (ldst_)[5] = p1_.y;                                \
        (ldst_)[6] = p1_.z; (ldst_)[7] = p1_.w;                                \
    } while (0)
#endif

#define LOAD_B(kb_)                                                            \
    do {                                                                       \
        const float* b0_ = B + (size_t)((kb_) * 32 + bk0) * N + tn * 128 + bn0;\
        r0a = ((const float4*)b0_)[0];                                         \
        r0b = ((const float4*)b0_)[1];                                         \
        r1a = ((const float4*)(b0_ + N))[0];                                   \
        r1b = ((const float4*)(b0_ + N))[1];                                   \
    } while (0)

#define STORE_B(buf_)                                                          \
    do {                                                                       \
        unsigned int* dst_ = &lB[buf_][bn0 * LDB_DW + kd];                     \
        dst_[0 * LDB_DW] = pack_bf16(r0a.x, r1a.x);                            \
        dst_[1 * LDB_DW] = pack_bf16(r0a.y, r1a.y);                            \
        dst_[2 * LDB_DW] = pack_bf16(r0a.z, r1a.z);                            \
        dst_[3 * LDB_DW] = pack_bf16(r0a.w, r1a.w);                            \
        dst_[4 * LDB_DW] = pack_bf16(r0b.x, r1b.x);                            \
        dst_[5 * LDB_DW] = pack_bf16(r0b.y, r1b.y);                            \
        dst_[6 * LDB_DW] = pack_bf16(r0b.z, r1b.z);                            \
        dst_[7 * LDB_DW] = pack_bf16(r0b.w, r1b.w);                            \
    } while (0)

    const int kIters = K >> 5;

    // Prologue: stage tile 0 into buffer 0.
    ISSUE_A(0, 0);
    LOAD_B(0);
    STORE_B(0);
    wait_async0();
    __syncthreads();

    for (int kb = 0; kb < kIters; ++kb) {
        const int cur = kb & 1;
        const int nxt = cur ^ 1;
        const bool has_next = (kb + 1) < kIters;

        if (has_next) {
            // Overlap next-tile movement with this tile's WMMAs.
            ISSUE_A(kb + 1, nxt);                       // async DMA -> LDS
            LOAD_B(kb + 1);                             // f32 weights -> regs
            if (kb + 2 < kIters)                        // warm GL2 two ahead
                __builtin_prefetch(B + (size_t)((kb + 2) * 32 + bk0) * N
                                   + tn * 128 + bn0, 0, 1);
        }

        Frag16 a0, a1, bf4[4];
        load_frag(lA[cur], wm,      LDA_DW, lane, a0.u);
        load_frag(lA[cur], wm + 16, LDA_DW, lane, a1.u);
        #pragma unroll
        for (int j = 0; j < 4; ++j)
            load_frag(lB[cur], wn + j * 16, LDB_DW, lane, bf4[j].u);

        #pragma unroll
        for (int j = 0; j < 4; ++j) {
            c[0][j] = __builtin_amdgcn_wmma_f32_16x16x32_bf16(
                false, a0.v, false, bf4[j].v, (short)0, c[0][j], false, false);
            c[1][j] = __builtin_amdgcn_wmma_f32_16x16x32_bf16(
                false, a1.v, false, bf4[j].v, (short)0, c[1][j], false, false);
        }

        if (has_next) STORE_B(nxt);   // convert + park next B tile in LDS

        wait_async0();
        __syncthreads();
    }

#undef ISSUE_A
#undef ASYNC_COPY
#undef LOAD_B
#undef STORE_B

    // Epilogue. C/D layout: n = lane&15; m = vgpr + (lane>=16 ? 8 : 0).
    const int nloc = lane & 15;
    const int mh   = ((lane >> 4) & 1) * 8;
    const size_t outbase = (size_t)e * CAPN * N;
    #pragma unroll
    for (int i = 0; i < 2; ++i) {
        const int rbase = tm * 128 + wm + i * 16 + mh;
        #pragma unroll
        for (int j = 0; j < 4; ++j) {
            const int n = tn * 128 + wn + j * 16 + nloc;
            const float bv = bias[n];
            #pragma unroll
            for (int vg = 0; vg < 8; ++vg) {
                float val = c[i][j][vg] + bv;
                const size_t off = outbase + (size_t)(rbase + vg) * N + n;
                if constexpr (RELU_BF16) {
                    val = val > 0.f ? val : 0.f;
                    ((unsigned short*)Outall)[off] = f2bf(val);
                } else {
                    ((float*)Outall)[off] = val;
                }
            }
        }
    }
}

// ---------------------------------------------------------------------------
// Combine: y = g1*keep1*eout[e1,c1] + g2*keep2*eout[e2,c2]
// ---------------------------------------------------------------------------
__global__ __launch_bounds__(256) void k_combine(
    const float* __restrict__ eout,
    const int* __restrict__ idx1, const int* __restrict__ idx2,
    const float* __restrict__ g1, const float* __restrict__ g2,
    const int* __restrict__ loc1, const int* __restrict__ loc2,
    float* __restrict__ y)
{
    const int t = blockIdx.x;
    const int col = threadIdx.x * 4;
    int e1 = idx1[t], l1 = loc1[t];
    int e2 = idx2[t], l2 = loc2[t];
    float w1 = (l1 < CAPN) ? g1[t] : 0.f;
    float w2 = (l2 < CAPN) ? g2[t] : 0.f;
    int c1 = l1 < (CAPN - 1) ? l1 : (CAPN - 1);
    int c2 = l2 < (CAPN - 1) ? l2 : (CAPN - 1);
    float4 a = *(const float4*)(eout + ((size_t)e1 * CAPN + c1) * MDIM + col);
    float4 b = *(const float4*)(eout + ((size_t)e2 * CAPN + c2) * MDIM + col);
    float4 o;
    o.x = w1 * a.x + w2 * b.x;
    o.y = w1 * a.y + w2 * b.y;
    o.z = w1 * a.z + w2 * b.z;
    o.w = w1 * a.w + w2 * b.w;
    *(float4*)(y + (size_t)t * MDIM + col) = o;
}

// ---------------------------------------------------------------------------
// Loss: l = sum_e me[e]*ce[e] * E/(S*S); deterministic serial reduce.
// ---------------------------------------------------------------------------
__global__ void k_loss(const float* __restrict__ partial_me,
                       const int* __restrict__ ce, float* __restrict__ out_loss)
{
    float me[NEXP] = {};
    for (int b = 0; b < S_TOK / 8; ++b)
        for (int e = 0; e < NEXP; ++e) me[e] += partial_me[b * NEXP + e];
    float l = 0.f;
    for (int e = 0; e < NEXP; ++e) l += me[e] * (float)ce[e];
    *out_loss = l * ((float)NEXP / ((float)S_TOK * (float)S_TOK));
}

// ---------------------------------------------------------------------------
extern "C" void kernel_launch(void* const* d_in, const int* in_sizes, int n_in,
                              void* d_out, int out_size, void* d_ws, size_t ws_size,
                              hipStream_t stream)
{
    (void)in_sizes; (void)n_in; (void)out_size; (void)ws_size;
    const float* x    = (const float*)d_in[0];
    const float* wg   = (const float*)d_in[1];
    const float* fc1w = (const float*)d_in[2];
    const float* fc1b = (const float*)d_in[3];
    const float* fc2w = (const float*)d_in[4];
    const float* fc2b = (const float*)d_in[5];
    float* y = (float*)d_out;

    char* ws = (char*)d_ws;
    size_t off = 0;
    auto alloc = [&](size_t bytes) -> char* {
        char* p = ws + off;
        off = (off + bytes + 255) & ~(size_t)255;
        return p;
    };
    int*   idx1 = (int*)  alloc((size_t)S_TOK * 4);
    int*   idx2 = (int*)  alloc((size_t)S_TOK * 4);
    float* g1   = (float*)alloc((size_t)S_TOK * 4);
    float* g2   = (float*)alloc((size_t)S_TOK * 4);
    int*   loc1 = (int*)  alloc((size_t)S_TOK * 4);
    int*   loc2 = (int*)  alloc((size_t)S_TOK * 4);
    float* pme  = (float*)alloc((size_t)(S_TOK / 8) * NEXP * 4);
    int*   ce   = (int*)  alloc(NEXP * 4);
    unsigned short* disp = (unsigned short*)alloc((size_t)NEXP * CAPN * MDIM * 2);
    unsigned short* hid  = (unsigned short*)alloc((size_t)NEXP * CAPN * HDIM * 2);
    float*          eout = (float*)         alloc((size_t)NEXP * CAPN * MDIM * 4);

    k_init<<<4096, 256, 0, stream>>>((unsigned int*)disp,
                                     (long)NEXP * CAPN * MDIM / 2);
    k_gate<<<S_TOK / 8, 256, 0, stream>>>(x, wg, idx1, idx2, g1, g2, pme);
    k_loc<<<1, 1, 0, stream>>>(idx1, idx2, loc1, loc2, ce);
    k_dispatch<<<S_TOK, 256, 0, stream>>>(x, idx1, idx2, loc1, loc2, disp);

    dim3 grid1((CAPN / 128) * (HDIM / 128), NEXP);
    k_gemm<true><<<grid1, 256, 0, stream>>>(disp, fc1w, fc1b, hid,
                                            MDIM, HDIM, HDIM / 128);
    dim3 grid2((CAPN / 128) * (MDIM / 128), NEXP);
    k_gemm<false><<<grid2, 256, 0, stream>>>(hid, fc2w, fc2b, eout,
                                             HDIM, MDIM, MDIM / 128);

    k_combine<<<S_TOK, 256, 0, stream>>>(eout, idx1, idx2, g1, g2, loc1, loc2, y);
    k_loss<<<1, 1, 0, stream>>>(pme, ce, y + (size_t)S_TOK * MDIM);
}
